// TransformerDecoderLayer_498216206917
// MI455X (gfx1250) — matrix operand order repaired and checked by reading, not compiled, not run on hardware
//
#include <hip/hip_runtime.h>
#include <hip/hip_bf16.h>

// ---------------------------------------------------------------------------
// CDNA5 (gfx1250, wave32) WMMA transformer decoder layer.
// GEMMs: v_wmma_f32_16x16x32_bf16, f32 accumulate.  Tile staging uses the
// gfx1250 async copy path (global_load_async_to_lds_b128 + s_wait_asynccnt),
// so the double-buffered LDS pipeline needs no staging VGPRs (fixes the
// scratch spills the register-staged version suffered under the 128-VGPR cap).
// Weights (transposed) and input activations are pre-converted once per
// launch to bf16; activations flow between GEMMs in bf16 via the epilogues.
// Only the attn@V GEMM converts f32->bf16 in-stage (attention weights are
// read exactly once; a pre-convert pass would cost more bandwidth).
// ---------------------------------------------------------------------------
typedef __attribute__((ext_vector_type(16))) __bf16 v16bf;
typedef __attribute__((ext_vector_type(8)))  float  v8f;

union FragBF { v16bf v; uint4 u[2]; };
union Pack8  { unsigned short s[8]; uint4 v; };

__device__ __forceinline__ unsigned short f2bf(float f) {
    union { float f; unsigned u; } x;
    x.f = f;
    unsigned u = x.u;
    unsigned r = 0x7FFFu + ((u >> 16) & 1u);   // round-to-nearest-even
    return (unsigned short)((u + r) >> 16);
}

// ---- CDNA5 async global->LDS copy (ASYNCcnt), emitted via inline asm ------
__device__ __forceinline__ void async_ld_b128(unsigned lds_off,
                                              const unsigned short* g) {
    asm volatile("global_load_async_to_lds_b128 %0, %1, off"
                 :: "v"(lds_off), "v"(g) : "memory");
}
__device__ __forceinline__ void wait_async0() {
    asm volatile("s_wait_asynccnt 0" ::: "memory");
}

// ---------------------------------------------------------------------------
// Flat convert: f32 -> bf16, 8 elements per thread (b128 in, b128 out)
// ---------------------------------------------------------------------------
__global__ __launch_bounds__(256)
void conv_bf16(const float* __restrict__ a, unsigned short* __restrict__ o)
{
    size_t i = (size_t)blockIdx.x * 256 + threadIdx.x;
    const float4* a4 = (const float4*)a;
    float4 f0 = a4[2 * i], f1 = a4[2 * i + 1];
    Pack8 pk;
    pk.s[0] = f2bf(f0.x); pk.s[1] = f2bf(f0.y);
    pk.s[2] = f2bf(f0.z); pk.s[3] = f2bf(f0.w);
    pk.s[4] = f2bf(f1.x); pk.s[5] = f2bf(f1.y);
    pk.s[6] = f2bf(f1.z); pk.s[7] = f2bf(f1.w);
    ((uint4*)o)[i] = pk.v;
}

// ---------------------------------------------------------------------------
// Weight convert + transpose:  W[k][n] (f32, KxN) -> Wt[n][k] (bf16, NxK)
// ---------------------------------------------------------------------------
__global__ __launch_bounds__(256)
void conv_transpose_bf16(const float* __restrict__ W,
                         unsigned short* __restrict__ Wt, int K, int N)
{
    __shared__ unsigned short t[32][33];
    const int n0 = blockIdx.x * 32, k0 = blockIdx.y * 32;
    const int c = threadIdx.x & 31, r = threadIdx.x >> 5;   // r: 0..7
    #pragma unroll
    for (int p = 0; p < 4; ++p) {
        int k = r + p * 8;
        t[k][c] = f2bf(W[(size_t)(k0 + k) * N + n0 + c]);
    }
    __syncthreads();
    #pragma unroll
    for (int p = 0; p < 4; ++p) {
        int n = r + p * 8;
        Wt[(size_t)(n0 + n) * K + k0 + c] = t[c][n];
    }
}

// ---------------------------------------------------------------------------
// Unified WMMA GEMM:  C[m,n] = act( A[m,:] @ B[:,n] + bias[n] )
//   A:  f32 (AF32) or bf16, row-major [M,K], lda; per-z offset b*sAb + h*sAh
//   Bt: bf16, element (k,n) at Bt[n*ldb + k]   (transposed-B layout, always)
//   Outputs (any subset): Cf f32 row-major;  Cb bf16 row-major, or
//   (OUTTRANS) Cb[col*ldcb + m] column-major (per-lane b128 stores).
// Block: 256 threads = 8 waves (4x2).  Tile 128 x BN, BK=32, double-buffered
// LDS (40KB), filled by async b128 copies overlapping the WMMA block.
// All problem dims divide tile dims (no predication; WMMA needs EXEC=all-1s).
// ---------------------------------------------------------------------------
template<int BN, bool AF32, bool RELU, bool OUTF32, bool OUTBF16, bool OUTTRANS>
__global__ __launch_bounds__(256, 2)
void gemm_wmma(const void* __restrict__ Ap, long lda, long sAb, long sAh,
               const unsigned short* __restrict__ Bt, long ldb, long sBb, long sBh,
               float* __restrict__ Cf, long ldc, long sCb, long sCh,
               unsigned short* __restrict__ Cb, long ldcb, long sCbb, long sCbh,
               const float* __restrict__ bias, int K, int H)
{
    constexpr int BM = 128, BK = 32, LD = 40;   // 40 = 32 + 8 pad (80B stride)
    __shared__ __attribute__((aligned(16))) unsigned short Ash[2][BM * LD];
    __shared__ __attribute__((aligned(16))) unsigned short Bsh[2][BN * LD];

    const int tid = threadIdx.x;
    const int z = blockIdx.z, bz = z / H, hz = z % H;
    const size_t aoff = (size_t)bz * sAb + (size_t)hz * sAh;
    Bt += (size_t)bz * sBb + (size_t)hz * sBh;
    if (OUTF32)  Cf += (size_t)bz * sCb  + (size_t)hz * sCh;
    if (OUTBF16) Cb += (size_t)bz * sCbb + (size_t)hz * sCbh;

    const int row0 = blockIdx.y * BM;
    const int col0 = blockIdx.x * BN;

    const int wave = tid >> 5, lane = tid & 31;
    const int wrow = wave >> 1;                 // 0..3 -> 32 rows each
    const int wcol = wave & 1;                  // 0..1 -> BN/2 cols each
    constexpr int WN = BN / 2;
    constexpr int TN = WN / 16;
    const int lm = lane & 15, lh = lane >> 4;

    // ---- per-thread staging addresses; advance by BK per stage ----
    const float*          afp = (const float*)Ap + aoff
                                + (size_t)(row0 + (tid >> 3)) * lda + (tid & 7) * 4;
    const unsigned short* ap  = (const unsigned short*)Ap + aoff
                                + (size_t)(row0 + (tid >> 2)) * lda + (tid & 3) * 8;
    const unsigned short* bp  = Bt + (size_t)(col0 + (tid >> 2)) * ldb + (tid & 3) * 8;

    // LDS destinations: byte offsets (low 32 bits of generic ptr == AS3 addr)
    unsigned short* asW = &Ash[0][(tid >> 3) * LD + (tid & 7) * 4];   // f32 path
    const unsigned asOff =
        (unsigned)(size_t)(const void*)&Ash[0][(tid >> 2) * LD + (tid & 3) * 8];
    const unsigned bsOff =
        (unsigned)(size_t)(const void*)&Bsh[0][(tid >> 2) * LD + (tid & 3) * 8];

    // fragment base pointers (per wave/lane)
    const unsigned short* aF = &Ash[0][(wrow * 32 + lm) * LD + lh * 8];
    const unsigned short* bF = &Bsh[0][(wcol * WN + lm) * LD + lh * 8];

    float4 arf[4];                               // A tile regs, f32 path only

    auto stageA = [&](int buf) {                 // bf16 A: async, no VGPRs
        unsigned d = asOff + (unsigned)buf * (BM * LD * 2);
        async_ld_b128(d, ap);
        async_ld_b128(d + 64 * LD * 2, ap + (size_t)64 * lda);
        ap += BK;
    };
    auto stageB = [&](int buf) {                 // B: async, no VGPRs
        unsigned d = bsOff + (unsigned)buf * (BN * LD * 2);
        #pragma unroll
        for (int p = 0; p < BN / 64; ++p)
            async_ld_b128(d + p * (64 * LD * 2), bp + (size_t)(p * 64) * ldb);
        bp += BK;
    };
    auto loadAf = [&]() {                        // f32 A: register staging
        #pragma unroll
        for (int p = 0; p < 4; ++p)
            arf[p] = *(const float4*)(afp + (size_t)(p * 32) * lda);
        afp += BK;
    };
    auto storeAf = [&](int buf) {                // convert + LDS store
        unsigned short* d = asW + buf * (BM * LD);
        #pragma unroll
        for (int p = 0; p < 4; ++p) {
            ushort4 u;
            u.x = f2bf(arf[p].x); u.y = f2bf(arf[p].y);
            u.z = f2bf(arf[p].z); u.w = f2bf(arf[p].w);
            *(ushort4*)(d + p * (32 * LD)) = u;
        }
    };

    v8f acc[2][TN];
    #pragma unroll
    for (int i = 0; i < 2; ++i)
        #pragma unroll
        for (int j = 0; j < TN; ++j)
            #pragma unroll
            for (int e = 0; e < 8; ++e) acc[i][j][e] = 0.0f;

    // prologue: fill buffer 0
    if constexpr (AF32) { loadAf(); storeAf(0); } else { stageA(0); }
    stageB(0);
    wait_async0();
    __syncthreads();

    const int nst = K / BK;
    for (int s = 0; s < nst; ++s) {
        const int cur = s & 1;
        const bool more = (s + 1 < nst);
        if (more) {                              // fill other buffer (async)
            if constexpr (AF32) loadAf(); else stageA(cur ^ 1);
            stageB(cur ^ 1);
        }

        const unsigned short* aC = aF + cur * (BM * LD);
        const unsigned short* bC = bF + cur * (BN * LD);

        // A fragment (ISA 7.12.2, 16-bit A 16x32): lanes 0-15 K=0-7|16-23,
        // lanes 16-31 K=8-15|24-31 -> two ds_load_b128.
        FragBF fa[2];
        #pragma unroll
        for (int tm = 0; tm < 2; ++tm) {
            fa[tm].u[0] = *(const uint4*)(aC + tm * (16 * LD));
            fa[tm].u[1] = *(const uint4*)(aC + tm * (16 * LD) + 16);
        }
        // B fragment (same K order as A): lanes 0-15 K=0-7|16-23,
        // lanes 16-31 K=8-15|24-31 -> two ds_load_b128.
        #pragma unroll
        for (int tn = 0; tn < TN; ++tn) {
            FragBF fb;
            fb.u[0] = *(const uint4*)(bC + tn * (16 * LD));
            fb.u[1] = *(const uint4*)(bC + tn * (16 * LD) + 16);
            #pragma unroll
            for (int tm = 0; tm < 2; ++tm) {
                acc[tm][tn] = __builtin_amdgcn_wmma_f32_16x16x32_bf16(
                    false, fa[tm].v, false, fb.v, (short)0, acc[tm][tn],
                    false, false);
            }
        }
        if (more) {
            if constexpr (AF32) storeAf(cur ^ 1);
            wait_async0();                       // async fills landed in LDS
        }
        __syncthreads();
    }

    // epilogue: C/D layout: lane n=lm, VGPR r -> m = r + 8*lh
    #pragma unroll
    for (int tm = 0; tm < 2; ++tm) {
        #pragma unroll
        for (int tn = 0; tn < TN; ++tn) {
            const int col = col0 + wcol * WN + tn * 16 + lm;
            const int mb  = row0 + wrow * 32 + tm * 16 + 8 * lh;
            const float bv = bias ? bias[col] : 0.0f;
            if constexpr (OUTF32) {
                #pragma unroll
                for (int r = 0; r < 8; ++r) {
                    float v = acc[tm][tn][r] + bv;
                    if (RELU) v = fmaxf(v, 0.0f);
                    Cf[(size_t)(mb + r) * ldc + col] = v;
                }
            }
            if constexpr (OUTBF16) {
                if constexpr (OUTTRANS) {
                    Pack8 pk;
                    #pragma unroll
                    for (int r = 0; r < 8; ++r) {
                        float v = acc[tm][tn][r] + bv;
                        if (RELU) v = fmaxf(v, 0.0f);
                        pk.s[r] = f2bf(v);
                    }
                    *(uint4*)&Cb[(size_t)col * ldcb + mb] = pk.v;  // 16B/lane
                } else {
                    #pragma unroll
                    for (int r = 0; r < 8; ++r) {
                        float v = acc[tm][tn][r] + bv;
                        if (RELU) v = fmaxf(v, 0.0f);
                        Cb[(size_t)(mb + r) * ldcb + col] = f2bf(v);
                    }
                }
            }
        }
    }
}

// ---------------------------------------------------------------------------
// In-place row softmax with pre-scale: p = softmax(p * 0.125), 1024 cols.
// Row cached in registers: single global read + single write (HBM floor).
// ---------------------------------------------------------------------------
__global__ __launch_bounds__(256)
void softmax_rows(float* __restrict__ w, float scale)
{
    __shared__ float red[256];
    const int tid = threadIdx.x;
    float* p = w + (size_t)blockIdx.x * 1024;

    float v[4];
    float m = -3.4e38f;
    #pragma unroll
    for (int j = 0; j < 4; ++j) {
        v[j] = p[tid + j * 256] * scale;
        m = fmaxf(m, v[j]);
    }
    red[tid] = m; __syncthreads();
    for (int s = 128; s > 0; s >>= 1) {
        if (tid < s) red[tid] = fmaxf(red[tid], red[tid + s]);
        __syncthreads();
    }
    m = red[0]; __syncthreads();

    float sum = 0.0f;
    #pragma unroll
    for (int j = 0; j < 4; ++j) {
        v[j] = __expf(v[j] - m);
        sum += v[j];
    }
    red[tid] = sum; __syncthreads();
    for (int s = 128; s > 0; s >>= 1) {
        if (tid < s) red[tid] += red[tid + s];
        __syncthreads();
    }
    float inv = 1.0f / red[0];
    #pragma unroll
    for (int j = 0; j < 4; ++j) p[tid + j * 256] = v[j] * inv;
}

// ---------------------------------------------------------------------------
// out = LayerNorm(a + rsd) * g + b ; also emits bf16 copy for next GEMM's A.
// ncols == 1024, 4 elems/thread.
// ---------------------------------------------------------------------------
__global__ __launch_bounds__(256)
void add_layernorm(const float* __restrict__ a, const float* __restrict__ rsd,
                   const float* __restrict__ g, const float* __restrict__ b,
                   float* __restrict__ out, unsigned short* __restrict__ outb)
{
    __shared__ float red[256];
    const int tid = threadIdx.x;
    const size_t base = (size_t)blockIdx.x * 1024;

    float v[4];
    float s = 0.0f;
    #pragma unroll
    for (int j = 0; j < 4; ++j) {
        int c = tid + j * 256;
        v[j] = a[base + c] + rsd[base + c];
        s += v[j];
    }
    red[tid] = s; __syncthreads();
    for (int st = 128; st > 0; st >>= 1) {
        if (tid < st) red[tid] += red[tid + st];
        __syncthreads();
    }
    float mu = red[0] * (1.0f / 1024.0f); __syncthreads();

    float vs = 0.0f;
    #pragma unroll
    for (int j = 0; j < 4; ++j) { float d = v[j] - mu; vs += d * d; }
    red[tid] = vs; __syncthreads();
    for (int st = 128; st > 0; st >>= 1) {
        if (tid < st) red[tid] += red[tid + st];
        __syncthreads();
    }
    float inv = rsqrtf(red[0] * (1.0f / 1024.0f) + 1e-6f);

    #pragma unroll
    for (int j = 0; j < 4; ++j) {
        int c = tid + j * 256;
        float y = (v[j] - mu) * inv * g[c] + b[c];
        out[base + c] = y;
        if (outb) outb[base + c] = f2bf(y);
    }
}

// ---------------------------------------------------------------------------
// Host-side orchestration
// ---------------------------------------------------------------------------
extern "C" void kernel_launch(void* const* d_in, const int* in_sizes, int n_in,
                              void* d_out, int out_size, void* d_ws, size_t ws_size,
                              hipStream_t stream)
{
    (void)in_sizes; (void)n_in; (void)out_size; (void)ws_size;

    const float* x   = (const float*)d_in[0];
    const float* enc = (const float*)d_in[1];
    const float* Wq1 = (const float*)d_in[2];  const float* bq1 = (const float*)d_in[3];
    const float* Wk1 = (const float*)d_in[4];  const float* bk1 = (const float*)d_in[5];
    const float* Wv1 = (const float*)d_in[6];  const float* bv1 = (const float*)d_in[7];
    const float* Wd1 = (const float*)d_in[8];  const float* bd1 = (const float*)d_in[9];
    const float* Wq2 = (const float*)d_in[10]; const float* bq2 = (const float*)d_in[11];
    const float* Wk2 = (const float*)d_in[12]; const float* bk2 = (const float*)d_in[13];
    const float* Wv2 = (const float*)d_in[14]; const float* bv2 = (const float*)d_in[15];
    const float* Wd2 = (const float*)d_in[16]; const float* bd2 = (const float*)d_in[17];
    const float* W1f = (const float*)d_in[18]; const float* b1f = (const float*)d_in[19];
    const float* W2f = (const float*)d_in[20]; const float* b2f = (const float*)d_in[21];
    const float* g1  = (const float*)d_in[22]; const float* be1 = (const float*)d_in[23];
    const float* g2  = (const float*)d_in[24]; const float* be2 = (const float*)d_in[25];
    const float* g3  = (const float*)d_in[26]; const float* be3 = (const float*)d_in[27];

    const int S = 1024, D = 1024, H = 16, DFF = 4096;
    const int M = 4 * S;                           // 4096
    const long SD = (long)S * D;                   // 1,048,576
    const long SS = (long)S * S;
    const size_t MD  = (size_t)M * D;              // 4M elements
    const size_t DD  = (size_t)D * D;              // 1M
    const size_t DF  = (size_t)D * DFF;            // 4M

    // ---- workspace layout ----
    float* densef = (float*)d_ws;                  // 4M f32
    float* out1f  = densef + MD;
    float* out2f  = out1f + MD;
    unsigned short* us = (unsigned short*)(out2f + MD);
    unsigned short* xb   = us;            us += MD;
    unsigned short* encb = us;            us += MD;
    unsigned short* wq1t = us;            us += DD;
    unsigned short* wk1t = us;            us += DD;
    unsigned short* wv1t = us;            us += DD;
    unsigned short* wd1t = us;            us += DD;
    unsigned short* wq2t = us;            us += DD;
    unsigned short* wk2t = us;            us += DD;
    unsigned short* wv2t = us;            us += DD;
    unsigned short* wd2t = us;            us += DD;
    unsigned short* w1ft = us;            us += DF;
    unsigned short* w2ft = us;            us += DF;
    unsigned short* qb   = us;            us += MD;
    unsigned short* kb   = us;            us += MD;
    unsigned short* vbT  = us;            us += MD;   // [D][M] col-major
    unsigned short* attnb= us;            us += MD;
    unsigned short* out1b= us;            us += MD;
    unsigned short* out2b= us;            us += MD;
    unsigned short* hffnb= us;            us += (size_t)M * DFF;

    float* out3 = (float*)d_out;
    float* w1   = out3 + MD;
    float* w2   = w1 + (size_t)4 * H * SS;

    dim3 blk(256);
    dim3 gProj(D / 128, M / 128, 1);
    dim3 gScores(S / 128, S / 128, 4 * H);
    dim3 gAV(1, S / 128, 4 * H);
    dim3 gFfn1(DFF / 128, M / 128, 1);
    dim3 gLN(M);
    dim3 gSm(4 * H * S);
    dim3 gConv((unsigned)(MD / 8 / 256));          // 2048 blocks

    // ---- pre-convert activations + weights to bf16 (once per launch) ----
    conv_bf16<<<gConv, blk, 0, stream>>>(x, xb);
    conv_bf16<<<gConv, blk, 0, stream>>>(enc, encb);
    dim3 gT(D / 32, D / 32);
    conv_transpose_bf16<<<gT, blk, 0, stream>>>(Wq1, wq1t, D, D);
    conv_transpose_bf16<<<gT, blk, 0, stream>>>(Wk1, wk1t, D, D);
    conv_transpose_bf16<<<gT, blk, 0, stream>>>(Wv1, wv1t, D, D);
    conv_transpose_bf16<<<gT, blk, 0, stream>>>(Wd1, wd1t, D, D);
    conv_transpose_bf16<<<gT, blk, 0, stream>>>(Wq2, wq2t, D, D);
    conv_transpose_bf16<<<gT, blk, 0, stream>>>(Wk2, wk2t, D, D);
    conv_transpose_bf16<<<gT, blk, 0, stream>>>(Wv2, wv2t, D, D);
    conv_transpose_bf16<<<gT, blk, 0, stream>>>(Wd2, wd2t, D, D);
    conv_transpose_bf16<<<dim3(DFF / 32, D / 32), blk, 0, stream>>>(W1f, w1ft, D, DFF);
    conv_transpose_bf16<<<dim3(D / 32, DFF / 32), blk, 0, stream>>>(W2f, w2ft, DFF, D);

    // ---------------- MHA1 (self-attention on x) ----------------
    gemm_wmma<128,false,false,false,true,false><<<gProj, blk, 0, stream>>>(
        xb,1024,0,0,  wq1t,1024,0,0,  nullptr,0,0,0,  qb,1024,0,0,  bq1, 1024, 1);
    gemm_wmma<128,false,false,false,true,false><<<gProj, blk, 0, stream>>>(
        xb,1024,0,0,  wk1t,1024,0,0,  nullptr,0,0,0,  kb,1024,0,0,  bk1, 1024, 1);
    gemm_wmma<128,false,false,false,true,true><<<gProj, blk, 0, stream>>>(
        xb,1024,0,0,  wv1t,1024,0,0,  nullptr,0,0,0,  vbT,(long)M,0,0,  bv1, 1024, 1);

    gemm_wmma<128,false,false,true,false,false><<<gScores, blk, 0, stream>>>(
        qb,1024, SD,64,   kb,1024, SD,64,
        w1,1024, (long)H*SS, SS,  nullptr,0,0,0,  nullptr, 64, H);
    softmax_rows<<<gSm, blk, 0, stream>>>(w1, 0.125f);

    gemm_wmma<64,true,false,false,true,false><<<gAV, blk, 0, stream>>>(
        w1,1024, (long)H*SS, SS,   vbT,(long)M, 1024, (long)64*M,
        nullptr,0,0,0,   attnb,1024, SD,64,  nullptr, S, H);

    gemm_wmma<128,false,false,true,false,false><<<gProj, blk, 0, stream>>>(
        attnb,1024,0,0,  wd1t,1024,0,0,  densef,1024,0,0,  nullptr,0,0,0, bd1, 1024, 1);
    add_layernorm<<<gLN, blk, 0, stream>>>(densef, x, g1, be1, out1f, out1b);

    // ---------------- MHA2 (cross-attention vs enc) ----------------
    gemm_wmma<128,false,false,false,true,false><<<gProj, blk, 0, stream>>>(
        out1b,1024,0,0,  wq2t,1024,0,0,  nullptr,0,0,0,  qb,1024,0,0,  bq2, 1024, 1);
    gemm_wmma<128,false,false,false,true,false><<<gProj, blk, 0, stream>>>(
        encb,1024,0,0,  wk2t,1024,0,0,  nullptr,0,0,0,  kb,1024,0,0,  bk2, 1024, 1);
    gemm_wmma<128,false,false,false,true,true><<<gProj, blk, 0, stream>>>(
        encb,1024,0,0,  wv2t,1024,0,0,  nullptr,0,0,0,  vbT,(long)M,0,0,  bv2, 1024, 1);

    gemm_wmma<128,false,false,true,false,false><<<gScores, blk, 0, stream>>>(
        qb,1024, SD,64,   kb,1024, SD,64,
        w2,1024, (long)H*SS, SS,  nullptr,0,0,0,  nullptr, 64, H);
    softmax_rows<<<gSm, blk, 0, stream>>>(w2, 0.125f);

    gemm_wmma<64,true,false,false,true,false><<<gAV, blk, 0, stream>>>(
        w2,1024, (long)H*SS, SS,   vbT,(long)M, 1024, (long)64*M,
        nullptr,0,0,0,   attnb,1024, SD,64,  nullptr, S, H);

    gemm_wmma<128,false,false,true,false,false><<<gProj, blk, 0, stream>>>(
        attnb,1024,0,0,  wd2t,1024,0,0,  densef,1024,0,0,  nullptr,0,0,0, bd2, 1024, 1);
    add_layernorm<<<gLN, blk, 0, stream>>>(densef, out1f, g2, be2, out2f, out2b);

    // ---------------- FFN ----------------
    gemm_wmma<128,false,true,false,true,false><<<gFfn1, blk, 0, stream>>>(
        out2b,1024,0,0,  w1ft,1024,0,0,  nullptr,0,0,0,
        hffnb,(long)DFF,0,0,  b1f, 1024, 1);
    gemm_wmma<128,false,false,true,false,false><<<gProj, blk, 0, stream>>>(
        hffnb,(long)DFF,0,0,  w2ft,(long)DFF,0,0,  densef,1024,0,0,
        nullptr,0,0,0,  b2f, DFF, 1);
    add_layernorm<<<gLN, blk, 0, stream>>>(densef, out2f, g3, be3, out3, nullptr);
}
